// EnhancedDualDomainJointModel_32315333935263
// MI455X (gfx1250) — matrix operand order repaired and checked by reading, not compiled, not run on hardware
//
#include <hip/hip_runtime.h>
#include <math.h>

// ---------------- model constants ----------------
#define BB   512      // batch
#define LL   128      // sequence length
#define DM   128      // d_model
#define DI   256      // d_inner
#define NS   16       // d_state
#define BCH  128      // batch chunk for mamba inner buffers
#define NCH  (BB/BCH)

__device__ __forceinline__ float geluf(float x)   { return 0.5f * x * (1.0f + erff(x * 0.70710678118654752f)); }
__device__ __forceinline__ float sigmf(float x)   { return 1.0f / (1.0f + expf(-x)); }
__device__ __forceinline__ float siluf(float x)   { return x / (1.0f + expf(-x)); }
__device__ __forceinline__ float softplusf(float x){ return x > 20.0f ? x : log1pf(expf(x)); }
#define BN_S 0.99999500003749971f

// ---------------- WMMA GEMM:  C[M,N] = A[M,K] * W[N,K]^T (+bias) ----------------
typedef __attribute__((ext_vector_type(16))) _Float16 v16h;
typedef __attribute__((ext_vector_type(8)))  float    v8f;

// block = 256 threads = 8 waves; each wave -> 16 rows x 64 cols; block tile 128x64.
// fp32 inputs converted to f16 fragments on load; fp32 accumulate via v_wmma_f32_16x16x32_f16.
__global__ void __launch_bounds__(256)
k_gemm_wmma(const float* __restrict__ A, const float* __restrict__ W,
            const float* __restrict__ bias, float* __restrict__ C,
            int M, int N, int K, int ldc)
{
    const int wave = threadIdx.x >> 5;
    const int lane = threadIdx.x & 31;
    const int half = lane >> 4;       // 0: lanes 0-15, 1: lanes 16-31
    const int lid  = lane & 15;
    const int m0 = blockIdx.x * 128 + wave * 16;
    const int n0 = blockIdx.y * 64;
    if (m0 >= M) return;              // wave-uniform exit (no barriers below)

    const v8f vzero = {0.f,0.f,0.f,0.f,0.f,0.f,0.f,0.f};
    v8f acc[4] = {vzero, vzero, vzero, vzero};

    const int kb  = half * 8;         // A K-offset interleave per ISA layout table
    const int row = m0 + lid;         // A row for this lane (both halves: same 16 rows)

    for (int k0 = 0; k0 < K; k0 += 32) {
        v16h a;
        const float* Ar = A + (size_t)row * K + k0;
#pragma unroll
        for (int j = 0; j < 8; j++) a[j]     = (_Float16)Ar[kb + j];
#pragma unroll
        for (int j = 0; j < 8; j++) a[8 + j] = (_Float16)Ar[kb + 16 + j];

#pragma unroll
        for (int t = 0; t < 4; t++) {
            int n = n0 + t * 16 + lid;
            // avoid control divergence: clamp row, zero via scale (EXEC stays all-1 for WMMA)
            float scl = (n < N) ? 1.0f : 0.0f;
            int   nn  = (n < N) ? n : 0;
            const float* Wr = W + (size_t)nn * K + k0 + kb; // B frag: same K interleave as A
            v16h bf;
#pragma unroll
            for (int j = 0; j < 16; j++) bf[j] = (_Float16)(Wr[j < 8 ? j : j + 8] * scl);
            acc[t] = __builtin_amdgcn_wmma_f32_16x16x32_f16(
                         false, a, false, bf, (short)0, acc[t], false, false);
        }
    }
#pragma unroll
    for (int t = 0; t < 4; t++) {
        int n = n0 + t * 16 + lid;
        if (n < N) {
            float bv = bias ? bias[n] : 0.0f;
#pragma unroll
            for (int r = 0; r < 8; r++) {
                int m = m0 + half * 8 + r;        // C/D layout: VGPR r -> M = r + half*8
                C[(size_t)m * ldc + n] = acc[t][r] + bv;
            }
        }
    }
}

// ---------------- row reduction helper (rows of 128) ----------------
__device__ __forceinline__ float rowsum128(float* red, int t, float v)
{
    red[t] = v; __syncthreads();
    for (int s = 64; s > 0; s >>= 1) { if (t < s) red[t] += red[t + s]; __syncthreads(); }
    float r = red[0]; __syncthreads();
    return r;
}

// ---------------- stage kernels ----------------
// temporal conv: tf[b,l,d] = gelu(BN_S*(conv7(x))) + pos[l,d]
__global__ void k_tpconv(const float* __restrict__ x, const float* __restrict__ w,
                         const float* __restrict__ bias, const float* __restrict__ pos,
                         float* __restrict__ tf)
{
    int r = blockIdx.x, d = threadIdx.x;          // r = b*L + l
    int b = r >> 7, l = r & 127;
    const float* xb = x + (size_t)b * 32 * 128;
    float s = bias[d];
    for (int c = 0; c < 32; c++) {
        const float* wr = w + (d * 32 + c) * 7;
        const float* xr = xb + c * 128;
        for (int k = 0; k < 7; k++) { int t = l - 3 + k; if ((unsigned)t < 128u) s += xr[t] * wr[k]; }
    }
    tf[(size_t)r * DM + d] = geluf(s * BN_S) + pos[l * DM + d];
}

// direct rDFT of x[:, :16, :nfft]: feat[b,c,k]=|F|, feat[b,16+c,k]=angle(F)
__global__ void k_dft_feat(const float* __restrict__ x, float* __restrict__ feat, int nfft, int F)
{
    int b = blockIdx.x, c = blockIdx.y, k = threadIdx.x;
    if (k >= F) return;
    const float* xr = x + ((size_t)b * 32 + c) * 128;
    float w = -6.283185307179586f * (float)k / (float)nfft;
    float re = 0.f, im = 0.f;
    for (int t = 0; t < nfft; t++) { float sn, cs; sincosf(w * (float)t, &sn, &cs); re += xr[t] * cs; im += xr[t] * sn; }
    feat[((size_t)b * 32 + c) * F + k]      = sqrtf(re * re + im * im);
    feat[((size_t)b * 32 + 16 + c) * F + k] = atan2f(im, re);
}

// conv1d(feat,[64,32,3],pad1)*BN_S -> gelu -> mean over freq -> h64[b,o]
__global__ void k_fpconv_mean(const float* __restrict__ feat, const float* __restrict__ cw,
                              const float* __restrict__ cb, float* __restrict__ h64, int F)
{
    int b = blockIdx.x, o = threadIdx.x;          // o < 64
    float accm = 0.f;
    for (int f = 0; f < F; f++) {
        float s = cb[o];
        for (int c = 0; c < 32; c++) {
            const float* fr = feat + ((size_t)b * 32 + c) * F;
            const float* wr = cw + (o * 32 + c) * 3;
            for (int k = 0; k < 3; k++) { int ff = f - 1 + k; if ((unsigned)ff < (unsigned)F) s += fr[ff] * wr[k]; }
        }
        accm += geluf(s * BN_S);
    }
    h64[(size_t)b * 64 + o] = accm / (float)F;
}

__global__ void k_fplin(const float* __restrict__ h64, const float* __restrict__ lw,
                        const float* __restrict__ lb, int dim, int off, float* __restrict__ comb)
{
    int b = blockIdx.x, j = threadIdx.x;
    if (j >= dim) return;
    float s = lb[j];
    for (int c = 0; c < 64; c++) s += h64[(size_t)b * 64 + c] * lw[j * 64 + c];
    comb[(size_t)b * DM + off + j] = s;
}

__global__ void k_dft_mag2(const float* __restrict__ x, float* __restrict__ mag2)
{
    int b = blockIdx.x, c = blockIdx.y, k = threadIdx.x;
    if (k >= 65) return;
    const float* xr = x + ((size_t)b * 32 + c) * 128;
    float w = -6.283185307179586f * (float)k / 128.0f;
    float re = 0.f, im = 0.f;
    for (int t = 0; t < 128; t++) { float sn, cs; sincosf(w * (float)t, &sn, &cs); re += xr[t] * cs; im += xr[t] * sn; }
    mag2[((size_t)b * 32 + c) * 65 + k] = re * re + im * im;
}

__global__ void k_band_avg(const float* __restrict__ mag2, float* __restrict__ bes)
{
    const int lo[5] = {1, 4, 8, 13, 30};
    const int hi[5] = {4, 8, 13, 30, 45};       // inclusive integer freq bins
    int b = blockIdx.x, j = threadIdx.x;        // j < 160
    int band = j >> 5, c = j & 31;
    float s = 0.f;
    for (int k = lo[band]; k <= hi[band]; k++) s += mag2[((size_t)b * 32 + c) * 65 + k];
    bes[(size_t)b * 160 + j] = s / (float)(hi[band] - lo[band] + 1);
}

__global__ void k_ln_gelu(float* __restrict__ buf, const float* __restrict__ g, const float* __restrict__ bt)
{
    __shared__ float red[128];
    int t = threadIdx.x; size_t r = blockIdx.x;
    float v = buf[r * DM + t];
    float mean = rowsum128(red, t, v) * (1.f / 128.f);
    float dv = v - mean;
    float var = rowsum128(red, t, dv * dv) * (1.f / 128.f);
    buf[r * DM + t] = geluf(dv * rsqrtf(var + 1e-5f) * g[t] + bt[t]);
}

__global__ void k_add_ln_out(const float* __restrict__ a, const float* __restrict__ b_,
                             const float* __restrict__ g, const float* __restrict__ bt,
                             float* __restrict__ out)
{
    __shared__ float red[128];
    int t = threadIdx.x; size_t r = blockIdx.x;
    float v = a[r * DM + t] + b_[r * DM + t];
    float mean = rowsum128(red, t, v) * (1.f / 128.f);
    float dv = v - mean;
    float var = rowsum128(red, t, dv * dv) * (1.f / 128.f);
    out[r * DM + t] = dv * rsqrtf(var + 1e-5f) * g[t] + bt[t];
}

__global__ void k_ln_inplace(float* __restrict__ buf, const float* __restrict__ g, const float* __restrict__ bt)
{
    __shared__ float red[128];
    int t = threadIdx.x; size_t r = blockIdx.x;
    float v = buf[r * DM + t];
    float mean = rowsum128(red, t, v) * (1.f / 128.f);
    float dv = v - mean;
    float var = rowsum128(red, t, dv * dv) * (1.f / 128.f);
    buf[r * DM + t] = dv * rsqrtf(var + 1e-5f) * g[t] + bt[t];
}

// tf = ln(tf + m*(1+sigmoid(gate[b]))) in-place (n2)
__global__ void k_gate_res_ln(float* __restrict__ tf, const float* __restrict__ m,
                              const float* __restrict__ gate,
                              const float* __restrict__ g, const float* __restrict__ bt)
{
    __shared__ float red[128];
    int t = threadIdx.x; size_t r = blockIdx.x; size_t b = r >> 7;
    float v = tf[r * DM + t] + m[r * DM + t] * (1.0f + sigmf(gate[b * DM + t]));
    float mean = rowsum128(red, t, v) * (1.f / 128.f);
    float dv = v - mean;
    float var = rowsum128(red, t, dv * dv) * (1.f / 128.f);
    tf[r * DM + t] = dv * rsqrtf(var + 1e-5f) * g[t] + bt[t];
}

// tf = ln(tf + add[b]) in-place (n3)
__global__ void k_add_bcast_ln(float* __restrict__ tf, const float* __restrict__ add,
                               const float* __restrict__ g, const float* __restrict__ bt)
{
    __shared__ float red[128];
    int t = threadIdx.x; size_t r = blockIdx.x; size_t b = r >> 7;
    float v = tf[r * DM + t] + add[b * DM + t];
    float mean = rowsum128(red, t, v) * (1.f / 128.f);
    float dv = v - mean;
    float var = rowsum128(red, t, dv * dv) * (1.f / 128.f);
    tf[r * DM + t] = dv * rsqrtf(var + 1e-5f) * g[t] + bt[t];
}

// causal depthwise conv (K=4, pad left 3) on xz[:, :256] then silu
__global__ void k_convsilu(const float* __restrict__ xz, const float* __restrict__ cw,
                           const float* __restrict__ cb, float* __restrict__ u)
{
    int r = blockIdx.x, d = threadIdx.x;          // r = lb*L + l, d < 256
    int l = r & 127;
    float s = cb[d];
    const float* wr = cw + d * 4;
    for (int k = 0; k < 4; k++) { int t = l - 3 + k; if (t >= 0) s += xz[(size_t)(r + t - l) * 512 + d] * wr[k]; }
    u[(size_t)r * DI + d] = siluf(s);
}

// dt = softplus(dbl[:, :8] @ dt_w^T + dt_b)
__global__ void k_dt(const float* __restrict__ dbl, const float* __restrict__ dt_w,
                     const float* __restrict__ dt_b, float* __restrict__ dt)
{
    int r = blockIdx.x, d = threadIdx.x;
    const float* dr = dbl + (size_t)r * 40;
    float s = dt_b[d];
    for (int j = 0; j < 8; j++) s += dr[j] * dt_w[d * 8 + j];
    dt[(size_t)r * DI + d] = softplusf(s);
}

// selective scan; fuses +u*D and *silu(z). One thread per (b,d), h[16] in regs.
__global__ void k_scan(const float* __restrict__ dtb, const float* __restrict__ u,
                       const float* __restrict__ dbl, const float* __restrict__ xz,
                       const float* __restrict__ A_log, const float* __restrict__ Dp,
                       float* __restrict__ y)
{
    int lb = blockIdx.x, d = threadIdx.x;
    float a[NS], h[NS];
#pragma unroll
    for (int n = 0; n < NS; n++) { a[n] = -expf(A_log[d * NS + n]); h[n] = 0.f; }
    float Dd = Dp[d];
    for (int l = 0; l < LL; l++) {
        size_t r = (size_t)lb * LL + l;
        float dt = dtb[r * DI + d];
        float uu = u[r * DI + d];
        float zz = xz[r * 512 + 256 + d];
        const float* Bm = dbl + r * 40 + 8;
        const float* Cm = Bm + 16;
        float du = dt * uu, acc = 0.f;
#pragma unroll
        for (int n = 0; n < NS; n++) {
            h[n] = h[n] * expf(dt * a[n]) + du * Bm[n];
            acc += h[n] * Cm[n];
        }
        acc += uu * Dd;
        y[r * DI + d] = acc * siluf(zz);
    }
}

__global__ void k_gelu(float* __restrict__ buf, int n)
{
    int i = blockIdx.x * blockDim.x + threadIdx.x;
    if (i < n) buf[i] = geluf(buf[i]);
}

// single-query MHA per batch element (4 heads, hd=32, L=128)
__global__ void k_attn(const float* __restrict__ qb, const float* __restrict__ kv,
                       float* __restrict__ o, int b0)
{
    __shared__ float sq[128];
    __shared__ float sa[4 * 128];
    __shared__ float sred[8];
    int t = threadIdx.x, lb = blockIdx.x, b = b0 + lb;
    sq[t] = qb[(size_t)b * DM + t];
    __syncthreads();
    const float* kr = kv + ((size_t)lb * LL + t) * 256;   // k row for l = t
    float s[4];
#pragma unroll
    for (int h = 0; h < 4; h++) {
        float acc = 0.f;
        for (int d = 0; d < 32; d++) acc += sq[h * 32 + d] * kr[h * 32 + d];
        s[h] = acc * 0.17677669529663687f;                // 1/sqrt(32)
        sa[h * 128 + t] = s[h];
    }
    __syncthreads();
    if (t < 4) {
        float mx = -1e30f;
        for (int l = 0; l < 128; l++) mx = fmaxf(mx, sa[t * 128 + l]);
        float sm = 0.f;
        for (int l = 0; l < 128; l++) sm += expf(sa[t * 128 + l] - mx);
        sred[t] = mx; sred[4 + t] = sm;
    }
    __syncthreads();
#pragma unroll
    for (int h = 0; h < 4; h++) sa[h * 128 + t] = expf(s[h] - sred[h]) / sred[4 + h];
    __syncthreads();
    int h = t >> 5, d = t & 31;
    float acc = 0.f;
    for (int l = 0; l < 128; l++)
        acc += sa[h * 128 + l] * kv[((size_t)lb * LL + l) * 256 + 128 + h * 32 + d];
    o[(size_t)b * DM + t] = acc;
}

// grouped temporal conv (groups=4, K=3, pad 1) + residual: out = tf + gelu(BN_S*conv)
__global__ void k_teconv(const float* __restrict__ tf, const float* __restrict__ w,
                         const float* __restrict__ bias, float* __restrict__ out)
{
    int r = blockIdx.x, o = threadIdx.x;
    int l = r & 127, g = o >> 5;
    float s = bias[o];
    for (int ic = 0; ic < 32; ic++) {
        const float* wr = w + (o * 32 + ic) * 3;
        int cin = g * 32 + ic;
        for (int k = 0; k < 3; k++) { int t = l - 1 + k; if ((unsigned)t < 128u) s += tf[(size_t)(r + t - l) * DM + cin] * wr[k]; }
    }
    out[(size_t)r * DM + o] = tf[(size_t)r * DM + o] + geluf(s * BN_S);
}

__global__ void k_gf(const float* __restrict__ tf, float* __restrict__ gf)
{
    int b = blockIdx.x, d = threadIdx.x;
    float acc = 0.f;
    for (int l = 0; l < LL; l++) acc += tf[((size_t)b * LL + l) * DM + d];
    gf[(size_t)b * DM + d] = acc * (1.f / (float)LL);
}

__global__ void k_logits(const float* __restrict__ gf, const float* __restrict__ cw,
                         const float* __restrict__ cb, float* __restrict__ out)
{
    int i = blockIdx.x * blockDim.x + threadIdx.x;
    if (i >= BB * 4) return;
    int b = i >> 2, j = i & 3;
    float s = cb[j];
    for (int d = 0; d < DM; d++) s += gf[(size_t)b * DM + d] * cw[j * DM + d];
    out[i] = s;
}

// ---------------- host ----------------
static inline void gemm(const float* A, const float* W, const float* bias, float* C,
                        int M, int N, int K, int ldc, hipStream_t s)
{
    dim3 grid((M + 127) / 128, (N + 63) / 64);
    k_gemm_wmma<<<grid, 256, 0, s>>>(A, W, bias, C, M, N, K, ldc);
}

extern "C" void kernel_launch(void* const* d_in, const int* in_sizes, int n_in,
                              void* d_out, int out_size, void* d_ws, size_t ws_size,
                              hipStream_t stream)
{
    (void)in_sizes; (void)n_in; (void)out_size; (void)ws_size;
    auto P = [&](int i) { return (const float*)d_in[i]; };

    const float* X     = P(0);
    const float* tp_w  = P(1);  const float* tp_b  = P(2);  const float* pos = P(3);
    const float* bd_w  = P(4);  const float* bd_b  = P(5);
    const float* bd_lg = P(6);  const float* bd_lb = P(7);
    const float* ff_g  = P(8);  const float* ff_b  = P(9);
    // 10..13: lc_w, lc_b, fn_g, fn_b -> dead code (logits use pre-ln pooled features)
    const float* cls_w = P(14); const float* cls_b = P(15);
    // fp groups at 16 + 4*i ; block i params at 28 + 25*i

    // workspace layout (floats), ~182 MB total
    float* w = (float*)d_ws; size_t off = 0;
    auto alloc = [&](size_t n) { float* p = w + off; off += n; return p; };
    float* tf    = alloc((size_t)BB * LL * DM);
    float* mbuf  = alloc((size_t)BB * LL * DM);
    float* ff    = alloc((size_t)BB * DM);
    float* comb  = alloc((size_t)BB * DM);
    float* band  = alloc((size_t)BB * DM);
    float* gate  = alloc((size_t)BB * DM);
    float* qb    = alloc((size_t)BB * DM);
    float* atto  = alloc((size_t)BB * DM);
    float* att2  = alloc((size_t)BB * DM);
    float* gfb   = alloc((size_t)BB * DM);
    float* gh    = alloc((size_t)BB * 256);
    float* h64   = alloc((size_t)BB * 64);
    float* feat  = alloc((size_t)BB * 32 * 65);
    float* mag2  = alloc((size_t)BB * 32 * 65);
    float* bes   = alloc((size_t)BB * 160);
    float* xz    = alloc((size_t)BCH * LL * 512);
    float* ub    = alloc((size_t)BCH * LL * DI);
    float* dbl   = alloc((size_t)BCH * LL * 40);
    float* dtb   = alloc((size_t)BCH * LL * DI);
    float* yb    = alloc((size_t)BCH * LL * DI);
    float* kv    = alloc((size_t)BCH * LL * 256);

    // stage 1: temporal conv + pos
    k_tpconv<<<dim3(BB * LL), 128, 0, stream>>>(X, tp_w, tp_b, pos, tf);

    // stage 2: frequency features -> ff[B,128]
    const int nffts[3] = {32, 64, 128};
    const int dims[3]  = {43, 43, 42};
    const int offs[3]  = {0, 43, 86};
    for (int i = 0; i < 3; i++) {
        int F = nffts[i] / 2 + 1;
        int bt = ((F + 31) / 32) * 32;
        k_dft_feat<<<dim3(BB, 16), bt, 0, stream>>>(X, feat, nffts[i], F);
        k_fpconv_mean<<<BB, 64, 0, stream>>>(feat, P(16 + 4 * i), P(17 + 4 * i), h64, F);
        k_fplin<<<BB, 64, 0, stream>>>(h64, P(18 + 4 * i), P(19 + 4 * i), dims[i], offs[i], comb);
    }
    k_dft_mag2<<<dim3(BB, 32), 96, 0, stream>>>(X, mag2);
    k_band_avg<<<BB, 160, 0, stream>>>(mag2, bes);
    gemm(bes, bd_w, bd_b, band, BB, DM, 160, DM, stream);
    k_ln_gelu<<<BB, 128, 0, stream>>>(band, bd_lg, bd_lb);
    k_add_ln_out<<<BB, 128, 0, stream>>>(comb, band, ff_g, ff_b, ff);

    // stage 3: 3 dual-domain blocks
    float* cur_tf = tf;
    float* cur_m  = mbuf;
    for (int ib = 0; ib < 3; ib++) {
        int base = 28 + 25 * ib;
        const float* in_w = P(base + 0),  *conv_w = P(base + 1),  *conv_b = P(base + 2);
        const float* x_w  = P(base + 3),  *dt_w   = P(base + 4),  *dt_b   = P(base + 5);
        const float* A_lg = P(base + 6),  *Dp     = P(base + 7),  *out_w  = P(base + 8);
        const float* g1_w = P(base + 9),  *g1_b   = P(base + 10);
        const float* g2_w = P(base + 11), *g2_b   = P(base + 12);
        const float* ai_w = P(base + 13), *ai_b   = P(base + 14);
        const float* ao_w = P(base + 15), *ao_b   = P(base + 16);
        const float* te_w = P(base + 17), *te_b   = P(base + 18);
        const float* n1g = P(base + 19), *n1b = P(base + 20);
        const float* n2g = P(base + 21), *n2b = P(base + 22);
        const float* n3g = P(base + 23), *n3b = P(base + 24);

        // mamba (chunked over B)
        for (int c = 0; c < NCH; c++) {
            const size_t co = (size_t)c * BCH * LL;
            gemm(cur_tf + co * DM, in_w, nullptr, xz, BCH * LL, 512, DM, 512, stream);
            k_convsilu<<<BCH * LL, 256, 0, stream>>>(xz, conv_w, conv_b, ub);
            gemm(ub, x_w, nullptr, dbl, BCH * LL, 40, DI, 40, stream);
            k_dt<<<BCH * LL, 256, 0, stream>>>(dbl, dt_w, dt_b, dtb);
            k_scan<<<BCH, 256, 0, stream>>>(dtb, ub, dbl, xz, A_lg, Dp, yb);
            gemm(yb, out_w, nullptr, cur_m + co * DM, BCH * LL, DM, DI, DM, stream);
        }
        k_ln_inplace<<<BB * LL, 128, 0, stream>>>(cur_m, n1g, n1b);

        // gate MLP on ff
        gemm(ff, g1_w, g1_b, gh, BB, 256, DM, 256, stream);
        k_gelu<<<(BB * 256 + 255) / 256, 256, 0, stream>>>(gh, BB * 256);
        gemm(gh, g2_w, g2_b, gate, BB, DM, 256, DM, stream);
        k_gate_res_ln<<<BB * LL, 128, 0, stream>>>(cur_tf, cur_m, gate, n2g, n2b);

        // attention: q from ff; k,v from sequence
        gemm(ff, ai_w, ai_b, qb, BB, DM, DM, DM, stream);
        for (int c = 0; c < NCH; c++) {
            const size_t co = (size_t)c * BCH * LL;
            gemm(cur_tf + co * DM, ai_w + 128 * 128, ai_b + 128, kv, BCH * LL, 256, DM, 256, stream);
            k_attn<<<BCH, 128, 0, stream>>>(qb, kv, atto, c * BCH);
        }
        gemm(atto, ao_w, ao_b, att2, BB, DM, DM, DM, stream);
        k_add_bcast_ln<<<BB * LL, 128, 0, stream>>>(cur_tf, att2, n3g, n3b);

        // grouped temporal conv + residual -> cur_m, then swap roles
        k_teconv<<<BB * LL, 128, 0, stream>>>(cur_tf, te_w, te_b, cur_m);
        float* tmp = cur_tf; cur_tf = cur_m; cur_m = tmp;
    }

    // final: global average pool -> logits (lc/fn are dead code for logits)
    k_gf<<<BB, 128, 0, stream>>>(cur_tf, gfb);
    k_logits<<<(BB * 4 + 127) / 128, 128, 0, stream>>>(gfb, cls_w, cls_b, (float*)d_out);
}
